// SVDObserver_81544249082280
// MI455X (gfx1250) — matrix operand order repaired and checked by reading, not compiled, not run on hardware
//
#include <hip/hip_runtime.h>
#include <hip/hip_bf16.h>

// ---------------------------------------------------------------------------
// Problem constants (match reference): B=16, C=256, H=W=128, K=24
// ---------------------------------------------------------------------------
#define Bc      16
#define Cc      256
#define HWc     16384        // 128*128
#define Kc      24
#define Np      32           // K padded to 32 for 16x16 WMMA tiling
#define SLICES  32           // pixel slices per batch
#define PIX_PER_WG (HWc / SLICES)   // 512
#define CHUNK   128          // pixels processed per inner stage
#define NCHUNK  (PIX_PER_WG / CHUNK) // 4
#define CS      32           // channel slice (WMMA K depth)
#define NCS     (Cc / CS)    // 8

typedef __attribute__((ext_vector_type(16))) __bf16 v16bf;
typedef __attribute__((ext_vector_type(8)))  float  v8f;

// fp32 -> bf16 (round-to-nearest-even), scalar — used only for W staging
__device__ __forceinline__ unsigned short f2bf(float f) {
  unsigned int u = __float_as_uint(f);
  unsigned int r = u + 0x7FFFu + ((u >> 16) & 1u);
  return (unsigned short)(r >> 16);
}

// two fp32 -> packed bf16x2 in ONE v_perm_b32 (+2 round-bias adds).
// result: [15:0] = bf16(lo), [31:16] = bf16(hi)
__device__ __forceinline__ unsigned pack2bf(float lo, float hi) {
  unsigned a = __float_as_uint(lo) + 0x8000u;   // round-half-up bias
  unsigned b = __float_as_uint(hi) + 0x8000u;
  // v_perm_b32: sel bytes 0-3 pick from S1(a), 4-7 from S0(b)
  return __builtin_amdgcn_perm(b, a, 0x07060302u);
}

union FragB16 {
  unsigned short u[16];
  uint4 q[2];
  v16bf v;
};

// ---------------------------------------------------------------------------
// Kernel 1: fused projection (H = X * W^T) + Gram (G += H^T H), bf16 WMMA.
// grid = (SLICES, B), block = 256 threads (8 wave32).
// Writes one 32x32 fp32 partial Gram per workgroup into d_ws.
// ---------------------------------------------------------------------------
__global__ __launch_bounds__(256) void gram_kernel(
    const float* __restrict__ x,       // (B, C, H*W)
    const float* __restrict__ Wc,      // (K, C)
    float* __restrict__ partial) {     // (B, SLICES, 32, 32)

  __shared__ unsigned short Wlds[Np][Cc];      // W^T padded, [n][c]  (16 KB)
  __shared__ unsigned short xlds[CHUNK][CS];   // x tile bf16 [pixel][c_local] (8 KB)
  __shared__ unsigned short Hlds[Np][CHUNK];   // H transposed [n][pixel] (8 KB)

  const int slice = blockIdx.x;
  const int b     = blockIdx.y;
  const int tid   = threadIdx.x;
  const int lane  = tid & 31;
  const int wave  = tid >> 5;
  const int l15   = lane & 15;
  const int halfSel = (lane < 16) ? 0 : 1;   // wave32 half select (ISA A/B layouts)

  // Stage W once: rows 24..31 zero-padded.
  for (int idx = tid; idx < Np * Cc; idx += 256) {
    int n = idx >> 8, c = idx & 255;
    float v = (n < Kc) ? Wc[n * Cc + c] : 0.0f;
    Wlds[n][c] = f2bf(v);
  }
  __syncthreads();

  const float* xb = x + (size_t)b * Cc * HWc;
  const int p0 = slice * PIX_PER_WG;

  v8f gacc = {};   // waves 0..3: one 16x16 tile of G each

  for (int ch = 0; ch < NCHUNK; ++ch) {
    const int pc = p0 + ch * CHUNK;
    v8f hacc0 = {}, hacc1 = {};   // per-wave H tile: 16 pixels x 32 outputs

    for (int cs = 0; cs < NCS; ++cs) {
      const int c0 = cs * CS;
      // ---- stage x slice: each thread loads 4 channel rows x 4 pixels ----
      {
        const int cl = (tid & 7) * 4;     // c_local 0..28
        const int pp = (tid >> 3) * 4;    // pixel   0..124
        const float4 r0 = *(const float4*)&xb[(size_t)(c0 + cl + 0) * HWc + pc + pp];
        const float4 r1 = *(const float4*)&xb[(size_t)(c0 + cl + 1) * HWc + pc + pp];
        const float4 r2 = *(const float4*)&xb[(size_t)(c0 + cl + 2) * HWc + pc + pp];
        const float4 r3 = *(const float4*)&xb[(size_t)(c0 + cl + 3) * HWc + pc + pp];
        if (cs + 1 < NCS)  // emits global_prefetch_b8 for next channel slice
          __builtin_prefetch(&xb[(size_t)(c0 + CS + cl) * HWc + pc + pp], 0, 1);
        uint2 w0, w1, w2, w3;
        w0.x = pack2bf(r0.x, r1.x);  w0.y = pack2bf(r2.x, r3.x);
        w1.x = pack2bf(r0.y, r1.y);  w1.y = pack2bf(r2.y, r3.y);
        w2.x = pack2bf(r0.z, r1.z);  w2.y = pack2bf(r2.z, r3.z);
        w3.x = pack2bf(r0.w, r1.w);  w3.y = pack2bf(r2.w, r3.w);
        *(uint2*)&xlds[pp + 0][cl] = w0;
        *(uint2*)&xlds[pp + 1][cl] = w1;
        *(uint2*)&xlds[pp + 2][cl] = w2;
        *(uint2*)&xlds[pp + 3][cl] = w3;
      }
      __syncthreads();

      // ---- GEMM1: H(16x32) += Xt(16x32) * Wt(32x32), two WMMAs per wave ----
      // A layout (bf16 16x32): lanes 0-15 row=l15 K=0..7,16..23; lanes 16-31 K=8..15,24..31
      FragB16 af;
      const int pixr = wave * 16 + l15;
      const int kbA  = halfSel * 8;
      af.q[0] = *(const uint4*)&xlds[pixr][kbA];
      af.q[1] = *(const uint4*)&xlds[pixr][kbA + 16];
      // B layout (bf16 32x16): lanes 0-15 col=l15 K=0..7,16..23; lanes 16-31 K=8..15,24..31
      FragB16 bf0, bf1;
      const int kbB = halfSel * 8;
      bf0.q[0] = *(const uint4*)&Wlds[l15][c0 + kbB];
      bf0.q[1] = *(const uint4*)&Wlds[l15][c0 + kbB + 16];
      bf1.q[0] = *(const uint4*)&Wlds[16 + l15][c0 + kbB];
      bf1.q[1] = *(const uint4*)&Wlds[16 + l15][c0 + kbB + 16];
      hacc0 = __builtin_amdgcn_wmma_f32_16x16x32_bf16(false, af.v, false, bf0.v,
                                                      (short)0, hacc0, false, false);
      hacc1 = __builtin_amdgcn_wmma_f32_16x16x32_bf16(false, af.v, false, bf1.v,
                                                      (short)0, hacc1, false, false);
      __syncthreads();
    }

    // ---- store H transposed into LDS: Hlds[n][pixel] (bf16, paired b32) ----
    {
      const int pixb = wave * 16 + halfSel * 8;  // D layout: lanes 16-31 hold M=8..15
      #pragma unroll
      for (int r = 0; r < 8; r += 2) {
        *(unsigned*)&Hlds[l15][pixb + r]      = pack2bf(hacc0[r], hacc0[r + 1]);
        *(unsigned*)&Hlds[16 + l15][pixb + r] = pack2bf(hacc1[r], hacc1[r + 1]);
      }
    }
    __syncthreads();

    // ---- GEMM2: G(32x32) += H^T(32x128) * H(128x32), waves 0..3 ----
    if (wave < 4) {
      const int mt = (wave >> 1) * 16;
      const int nt = (wave & 1) * 16;
      #pragma unroll
      for (int ps = 0; ps < CHUNK / 32; ++ps) {
        FragB16 a2, b2;
        const int pbA = ps * 32 + halfSel * 8;
        a2.q[0] = *(const uint4*)&Hlds[mt + l15][pbA];
        a2.q[1] = *(const uint4*)&Hlds[mt + l15][pbA + 16];
        const int pbB = ps * 32 + halfSel * 8;
        b2.q[0] = *(const uint4*)&Hlds[nt + l15][pbB];
        b2.q[1] = *(const uint4*)&Hlds[nt + l15][pbB + 16];
        gacc = __builtin_amdgcn_wmma_f32_16x16x32_bf16(false, a2.v, false, b2.v,
                                                       (short)0, gacc, false, false);
      }
    }
    __syncthreads();
  }

  // ---- write partial 32x32 Gram (fp32) ----
  if (wave < 4) {
    float* pout = partial + (size_t)(b * SLICES + slice) * (Np * Np);
    const int mt = (wave >> 1) * 16;
    const int nt = (wave & 1) * 16;
    const int rowOff = halfSel * 8;
    #pragma unroll
    for (int r = 0; r < 8; ++r)
      pout[(mt + rowOff + r) * Np + (nt + l15)] = gacc[r];
  }
}

// ---------------------------------------------------------------------------
// Kernel 2: reduce partials, Jacobi eigh on 24x24, features. 1 wave / batch.
// Output layout: [S (B,K) | Vh (B,K,K) | feats (B,2K+2) | novelty (B,K)]
// ---------------------------------------------------------------------------
__device__ __forceinline__ bool finitef(float v) { return __builtin_isfinite(v); }

__global__ __launch_bounds__(32) void eig_kernel(
    const float* __restrict__ partial,
    const float* __restrict__ ema,
    float* __restrict__ out) {

  __shared__ float Ae[Kc][Kc + 1];
  __shared__ float Ve[Kc][Kc + 1];
  __shared__ float sArr[Kc];
  __shared__ float snorm[Kc];
  __shared__ int   perm[Kc];
  __shared__ float red[32], red2[32];

  const int b = blockIdx.x;
  const int t = threadIdx.x;

  // reduce 32 partial Grams -> Ae ; Ve = I
  for (int e = t; e < Kc * Kc; e += 32) {
    const int r = e / Kc, c = e % Kc;
    float s = 0.f;
    for (int sl = 0; sl < SLICES; ++sl)
      s += partial[(size_t)(b * SLICES + sl) * (Np * Np) + r * Np + c];
    Ae[r][c] = s;
    Ve[r][c] = (r == c) ? 1.f : 0.f;
  }
  __syncthreads();

  // cyclic Jacobi sweeps
  for (int sweep = 0; sweep < 12; ++sweep) {
    for (int p = 0; p < Kc - 1; ++p) {
      for (int q = p + 1; q < Kc; ++q) {
        const float apq = Ae[p][q];
        const float app = Ae[p][p];
        const float aqq = Ae[q][q];
        __syncthreads();
        if (fabsf(apq) > 1e-12f * (fabsf(app) + fabsf(aqq)) + 1e-30f) {  // uniform
          const float theta = 0.5f * (aqq - app) / apq;
          const float tt = (theta >= 0.f ? 1.f : -1.f) /
                           (fabsf(theta) + sqrtf(1.f + theta * theta));
          const float cc = rsqrtf(1.f + tt * tt);
          const float ss = tt * cc;
          if (t < Kc) {               // row rotation: A <- J^T A
            const float apj = Ae[p][t], aqj = Ae[q][t];
            Ae[p][t] = cc * apj - ss * aqj;
            Ae[q][t] = ss * apj + cc * aqj;
          }
          __syncthreads();
          if (t < Kc) {               // column rotation: A <- A J ; V <- V J
            const float ajp = Ae[t][p], ajq = Ae[t][q];
            Ae[t][p] = cc * ajp - ss * ajq;
            Ae[t][q] = ss * ajp + cc * ajq;
            const float vjp = Ve[t][p], vjq = Ve[t][q];
            Ve[t][p] = cc * vjp - ss * vjq;
            Ve[t][q] = ss * vjp + cc * vjq;
          }
          __syncthreads();
        }
      }
    }
  }

  // descending sort (matches eigh ascending + [::-1])
  if (t < Kc) {
    const float wi = Ae[t][t];
    int rank = 0;
    for (int j = 0; j < Kc; ++j) {
      const float wj = Ae[j][j];
      rank += (wj > wi) || (wj == wi && j < t);
    }
    perm[rank] = t;
  }
  __syncthreads();

  if (t < Kc) {
    const int src = perm[t];
    float s = sqrtf(fmaxf(Ae[src][src], 0.f));
    s = fmaxf(s, 1e-6f);
    if (!finitef(s)) s = 1.f;
    sArr[t] = s;
  }
  __syncthreads();

  float ssum = 0.f;
  for (int j = 0; j < Kc; ++j) ssum += sArr[j];

  float* Sout = out;                                    // [0, 384)
  float* Vout = out + Bc * Kc;                          // [384, 9600)
  float* Fout = out + Bc * Kc + Bc * Kc * Kc;           // [9600, 10400)
  float* Nout = Fout + Bc * (2 * Kc + 2);               // [10400, 10784)

  if (t < Kc) {
    const float s = sArr[t];
    Sout[b * Kc + t] = s;
    Nout[b * Kc + t] = s - ema[t];
    float sn = s / (ssum + 1e-8f);
    if (!finitef(sn)) sn = 0.f;
    snorm[t] = sn;
    Fout[b * (2 * Kc + 2) + t] = sn;
    float vd = Ve[t][perm[t]];
    if (!finitef(vd)) vd = 0.f;
    Fout[b * (2 * Kc + 2) + Kc + t] = vd;
  }
  __syncthreads();

  // Vh = (sorted V)^T, plus Frobenius sums for off-diagonal energy
  float sq = 0.f, dq = 0.f;
  for (int e = t; e < Kc * Kc; e += 32) {
    const int i = e / Kc, j = e % Kc;
    float v = Ve[j][perm[i]];
    if (!finitef(v)) v = 0.f;
    Vout[b * Kc * Kc + e] = v;
    sq += v * v;
    if (i == j) dq += v * v;
  }
  red[t] = sq;
  red2[t] = dq;
  __syncthreads();

  if (t == 0) {
    float S2 = 0.f, D2 = 0.f;
    for (int j = 0; j < 32; ++j) { S2 += red[j]; D2 += red2[j]; }
    float off = fmaxf(S2 - D2, 0.f);
    if (!finitef(off)) off = 0.f;
    Fout[b * (2 * Kc + 2) + 2 * Kc] = off;
    float ent = 0.f;
    for (int j = 0; j < Kc; ++j) {
      const float sn = snorm[j];
      ent -= sn * logf(fmaxf(sn, 1e-8f));
    }
    if (!finitef(ent)) ent = 0.f;
    Fout[b * (2 * Kc + 2) + 2 * Kc + 1] = ent;
  }
}

// ---------------------------------------------------------------------------
extern "C" void kernel_launch(void* const* d_in, const int* in_sizes, int n_in,
                              void* d_out, int out_size, void* d_ws, size_t ws_size,
                              hipStream_t stream) {
  (void)in_sizes; (void)n_in; (void)out_size; (void)ws_size;
  const float* x   = (const float*)d_in[0];   // (B,C,H,W) fp32
  const float* Wc  = (const float*)d_in[1];   // (K,C)     fp32
  const float* ema = (const float*)d_in[2];   // (K,)      fp32
  float* out       = (float*)d_out;
  float* partials  = (float*)d_ws;            // needs B*SLICES*32*32*4 = 2 MiB

  dim3 grid(SLICES, Bc);
  gram_kernel<<<grid, 256, 0, stream>>>(x, Wc, partials);
  eig_kernel<<<Bc, 32, 0, stream>>>(partials, ema, out);
}
